// Monarch_56788057588462
// MI455X (gfx1250) — matrix operand order repaired and checked by reading, not compiled, not run on hardware
//
#include <hip/hip_runtime.h>
#include <stdint.h>

typedef __attribute__((ext_vector_type(16))) __bf16         v16bf;
typedef __attribute__((ext_vector_type(8)))  __bf16         v8bf;
typedef __attribute__((ext_vector_type(2)))  __bf16         v2bf;
typedef __attribute__((ext_vector_type(8)))  float          v8f;
typedef __attribute__((ext_vector_type(2)))  float          v2f;
typedef __attribute__((ext_vector_type(16))) unsigned short v16u;
typedef __attribute__((ext_vector_type(8)))  unsigned short v8u;

#define IN_DIM   4096
#define NB       64
#define ROWS     16
#define HSTRIDE  72                     // ushorts per h row (64 + 8 pad, keeps 16B align)
#define HSLICE   (ROWS * HSTRIDE + 8)   // 1160 ushorts per k2 slice (bank-spread pad)
#define HSIZE    (64 * HSLICE)          // 74240 ushorts = 148480 B
#define OROW     1096                   // floats per out-staging row (64*17 + 8)
#define OSIZE    (ROWS * OROW)          // 17536 floats = 70144 B
#define SMEM_BYTES (HSIZE * 2 + OSIZE * 4)   // 218,624 B < 320 KB

// Native f32 -> bf16 (RNE) conversions: let LLVM lower fptrunc-to-bfloat to the
// hardware packed convert on gfx1250 instead of hand-rolled bit twiddling.
__device__ __forceinline__ unsigned short f2bf(float f) {
  return __builtin_bit_cast(unsigned short, (__bf16)f);
}

__device__ __forceinline__ unsigned int pk2bf(float a, float b) {
  v2f t; t[0] = a; t[1] = b;
  v2bf p = __builtin_convertvector(t, v2bf);
  return __builtin_bit_cast(unsigned int, p);
}

__device__ __forceinline__ v8u cvt8(const float* p) {
  v8f f = *(const v8f*)p;                       // 32B load (2x b128)
  v8bf b = __builtin_convertvector(f, v8bf);
  return __builtin_bit_cast(v8u, b);
}

__device__ __forceinline__ v16bf mk16(v8u lo, v8u h8) {
  v16u u = __builtin_shufflevector(lo, h8, 0,1,2,3,4,5,6,7,8,9,10,11,12,13,14,15);
  return __builtin_bit_cast(v16bf, u);
}

// A operand from LDS h-slice: a[0..7] = base[off .. off+7], a[8..15] = base[off+8 .. off+15]
__device__ __forceinline__ v16bf ldsA(const unsigned short* base, int off) {
  v8u lo = *(const v8u*)(base + off);
  v8u h8 = *(const v8u*)(base + off + 8);
  return mk16(lo, h8);
}

// Convert + block-transpose weights: wt[k][o][i] = bf16(w[k][i][o]); 64x64x64 each.
__global__ __launch_bounds__(256) void monarch_wt(const float* __restrict__ w,
                                                  unsigned short* __restrict__ wt) {
  int idx = blockIdx.x * 256 + threadIdx.x;     // 65536 threads, 4 elems each
#pragma unroll
  for (int e = 0; e < 4; ++e) {
    int id = idx + e * 65536;
    int k = id >> 12, rem = id & 4095, i = rem >> 6, o = rem & 63;
    wt[(k << 12) + (o << 6) + i] = f2bf(w[id]);
  }
}

__global__ __launch_bounds__(256) void monarch_main(
    const float* __restrict__ X, const unsigned short* __restrict__ wt1,
    const unsigned short* __restrict__ wt2, const float* __restrict__ bias,
    float* __restrict__ out)
{
  extern __shared__ char smem[];
  unsigned short* hsm = (unsigned short*)smem;            // h[k2][row][i] (bf16, padded)
  float*          osm = (float*)(smem + HSIZE * 2);       // out staging (swizzled)

  const int lane = threadIdx.x & 31;
  const int wave = threadIdx.x >> 5;
  const int n16  = lane & 15;
  const int hi   = (lane >> 4) & 1;
  const int sh   = hi ? 8 : 0;         // K sub-offset for A operand (wave32 halves)
  const int rowbase = blockIdx.x * ROWS;

  // ---------------- stage 1: h[b,i,k2] = X_block_i @ w1[i], stored transposed ----------------
  for (int blk = wave; blk < NB; blk += 8) {
    const float* xp = X + (size_t)(rowbase + n16) * IN_DIM + blk * 64;
    v16bf a0 = mk16(cvt8(xp + 2 * sh),      cvt8(xp + 8 + 2 * sh));   // K = 0..31
    v16bf a1 = mk16(cvt8(xp + 32 + 2 * sh), cvt8(xp + 40 + 2 * sh));   // K = 32..63
#pragma unroll
    for (int nt = 0; nt < 4; ++nt) {
      const unsigned short* wp = wt1 + (blk << 12) + ((nt * 16 + n16) << 6) + hi * 16;
      v16bf b0 = __builtin_bit_cast(v16bf, *(const v16u*)wp);
      v16bf b1 = __builtin_bit_cast(v16bf, *(const v16u*)(wp + 32));
      v8f c = {};
      c = __builtin_amdgcn_wmma_f32_16x16x32_bf16(false, a0, false, b0, (short)0, c, false, false);
      c = __builtin_amdgcn_wmma_f32_16x16x32_bf16(false, a1, false, b1, (short)0, c, false, false);
      // lane holds column k2 = nt*16+n16; vgpr r holds row r + 8*hi; write h[k2][row][blk]
      unsigned short* hp = hsm + (nt * 16 + n16) * HSLICE + (hi * 8) * HSTRIDE + blk;
#pragma unroll
      for (int r = 0; r < 8; r += 2) {
        unsigned int p = pk2bf(c[r], c[r + 1]);
        hp[(r + 0) * HSTRIDE] = (unsigned short)p;
        hp[(r + 1) * HSTRIDE] = (unsigned short)(p >> 16);
      }
    }
  }
  __syncthreads();

  // ---------------- stage 2: o[b,k2,o2] = h2[b,k2,:] @ w2[k2], coalesced flush ----------------
  for (int nt = 0; nt < 4; ++nt) {
    for (int k2 = wave; k2 < 64; k2 += 8) {
      const unsigned short* hp = hsm + k2 * HSLICE + n16 * HSTRIDE;
      v16bf a0 = ldsA(hp, 2 * sh);           // K = 0..31  (i dimension)
      v16bf a1 = ldsA(hp, 32 + 2 * sh);      // K = 32..63
      const unsigned short* wp = wt2 + (k2 << 12) + ((nt * 16 + n16) << 6) + hi * 16;
      v16bf b0 = __builtin_bit_cast(v16bf, *(const v16u*)wp);
      v16bf b1 = __builtin_bit_cast(v16bf, *(const v16u*)(wp + 32));
      v8f c = {};
      c = __builtin_amdgcn_wmma_f32_16x16x32_bf16(false, a0, false, b0, (short)0, c, false, false);
      c = __builtin_amdgcn_wmma_f32_16x16x32_bf16(false, a1, false, b1, (short)0, c, false, false);
      // stage into LDS, swizzled col = k2*17 + o2l to spread banks
      float* op = osm + (hi * 8) * OROW + k2 * 17 + n16;
#pragma unroll
      for (int r = 0; r < 8; ++r) op[r * OROW] = c[r];
    }
    __syncthreads();
    {
      // flush 16 rows x 1024 cols of this n-tile, bias added, float4 coalesced stores
      const int t   = threadIdx.x;        // 0..255, 4 columns each
      const int jl  = 4 * t;              // jl = o2l*64 + k2
      const int k2f = jl & 63, o2f = jl >> 6;
      const float4 bv = *(const float4*)(bias + nt * 1024 + jl);
      float* og = out + (size_t)rowbase * 4096 + nt * 1024 + jl;
      const float* rp0 = osm + k2f * 17 + o2f;
#pragma unroll 4
      for (int row = 0; row < ROWS; ++row) {
        const float* rp = rp0 + row * OROW;
        float4 v;
        v.x = rp[0]  + bv.x;
        v.y = rp[17] + bv.y;
        v.z = rp[34] + bv.z;
        v.w = rp[51] + bv.w;
        *(float4*)(og + (size_t)row * 4096) = v;
      }
    }
    __syncthreads();
  }
}

extern "C" void kernel_launch(void* const* d_in, const int* in_sizes, int n_in,
                              void* d_out, int out_size, void* d_ws, size_t ws_size,
                              hipStream_t stream) {
  const float* x    = (const float*)d_in[0];
  const float* w1   = (const float*)d_in[1];
  const float* w2   = (const float*)d_in[2];
  const float* bias = (const float*)d_in[3];
  float* out = (float*)d_out;

  unsigned short* wt1 = (unsigned short*)d_ws;          // 64*64*64 bf16 = 512 KB
  unsigned short* wt2 = wt1 + 64 * 64 * 64;             // another 512 KB

  (void)hipFuncSetAttribute((const void*)monarch_main,
                            hipFuncAttributeMaxDynamicSharedMemorySize, SMEM_BYTES);

  monarch_wt<<<256, 256, 0, stream>>>(w1, wt1);
  monarch_wt<<<256, 256, 0, stream>>>(w2, wt2);

  const int n_rows = 8 * 2048;                          // 16384
  monarch_main<<<n_rows / ROWS, 256, SMEM_BYTES, stream>>>(x, wt1, wt2, bias, out);
}
